// PolarConv_25546465477063
// MI455X (gfx1250) — matrix operand order repaired and checked by reading, not compiled, not run on hardware
//
#include <hip/hip_runtime.h>
#include <hip/hip_bf16.h>

typedef __attribute__((ext_vector_type(16))) _Float16 v16h;
typedef __attribute__((ext_vector_type(8)))  float    v8f;

#define NWAVES_PER_BLOCK 8
#define IN_DIM 16
#define H_DIM  32
#define OUT_DIM 16
#define W3_COLS 256   // IN_DIM * OUT_DIM

union ABV { int4 q[2]; v16h v; };
union PK16 { _Float16 h[16]; int4 q[2]; };

__device__ __forceinline__ void wave_lds_fence() {
    // CDNA5 split counters: wait for all outstanding LDS ops of this wave.
    asm volatile("s_wait_dscnt 0" ::: "memory");
}

__global__ __launch_bounds__(256) void polarconv_kernel(
    const float* __restrict__ feats,      // [N,16]
    const float* __restrict__ xyz,        // [N,3]
    const int*   __restrict__ nidx,       // [E]
    const int*   __restrict__ rsplits,    // [N+1]
    const float* __restrict__ ndist,      // [E]
    const float* __restrict__ W1,         // [4,32]
    const float* __restrict__ b1,         // [32]
    const float* __restrict__ W2,         // [32,32]
    const float* __restrict__ b2,         // [32]
    const float* __restrict__ W3,         // [32,256]
    const float* __restrict__ b3,         // [256]
    float*       __restrict__ out,        // [N,16]
    int nq)
{
    // ---- block-shared, built once: f16 transposed weights + f32 biases ----
    __shared__ __align__(16) _Float16 sW2T[H_DIM * H_DIM];     // [col][k]
    __shared__ __align__(16) _Float16 sW3T[W3_COLS * H_DIM];   // [col][k]
    __shared__ float sW1[4 * H_DIM];
    __shared__ float sb1[H_DIM], sb2[H_DIM], sb3[W3_COLS];
    // ---- per-wave scratch ----
    __shared__ __align__(16) _Float16 sH[NWAVES_PER_BLOCK][16 * H_DIM];  // h1 then h2, row-major [edge][32]
    __shared__ __align__(16) float    sF[NWAVES_PER_BLOCK][16 * IN_DIM]; // gathered feats [edge][16]

    const int tid = threadIdx.x;
    for (int idx = tid; idx < H_DIM * H_DIM; idx += blockDim.x) {
        int c = idx >> 5, k = idx & 31;
        sW2T[c * H_DIM + k] = (_Float16)W2[k * H_DIM + c];
    }
    for (int idx = tid; idx < W3_COLS * H_DIM; idx += blockDim.x) {
        int c = idx >> 5, k = idx & 31;
        sW3T[c * H_DIM + k] = (_Float16)W3[k * W3_COLS + c];
    }
    for (int idx = tid; idx < 4 * H_DIM; idx += blockDim.x) sW1[idx] = W1[idx];
    for (int idx = tid; idx < H_DIM; idx += blockDim.x) { sb1[idx] = b1[idx]; sb2[idx] = b2[idx]; }
    for (int idx = tid; idx < W3_COLS; idx += blockDim.x) sb3[idx] = b3[idx];
    __syncthreads();

    const int wid  = tid >> 5;
    const int lane = tid & 31;
    const int hi   = lane >> 4;   // which wave half
    const int ln   = lane & 15;

    _Float16* hbase = &sH[wid][0];
    float*    fbase = &sF[wid][0];

    const int gwave   = blockIdx.x * NWAVES_PER_BLOCK + wid;
    const int nwavesT = gridDim.x * NWAVES_PER_BLOCK;

    for (int q = gwave; q < nq; q += nwavesT) {
        const int base = rsplits[q];
        const int end  = rsplits[q + 1];
        const float qx = xyz[q * 3 + 0];
        const float qy = xyz[q * 3 + 1];
        const float qz = xyz[q * 3 + 2];

        float acc[8];
#pragma unroll
        for (int p = 0; p < 8; ++p) acc[p] = 0.f;

        for (int c0 = base; c0 < end; c0 += 16) {
            const int  e     = c0 + ln;
            const bool valid = (e < end);
            const int  nbr   = valid ? nidx[e] : 0;
            const float d    = valid ? ndist[e] : 0.f;
            const float r    = __builtin_sqrtf(d + 1e-7f);
            const float inv  = 1.0f / r;
            float dx = 0.f, dy = 0.f, dz = 0.f;
            if (valid) {
                dx = xyz[nbr * 3 + 0] - qx;
                dy = xyz[nbr * 3 + 1] - qy;
                dz = xyz[nbr * 3 + 2] - qz;
            }
            const float p0 = r, p1 = dx * inv, p2 = dz * inv, p3 = dy * inv;

            // gather neighbor feats: lane covers channels hi*8..hi*8+7 of edge ln
            float4 f0 = make_float4(0.f, 0.f, 0.f, 0.f), f1 = f0;
            if (valid) {
                const float* fr = feats + (size_t)nbr * IN_DIM + hi * 8;
                f0 = *(const float4*)fr;
                f1 = *(const float4*)(fr + 4);
            }
            {
                float4* fd = (float4*)&fbase[ln * IN_DIM + hi * 8];
                fd[0] = f0; fd[1] = f1;
            }

            // ---- layer 1 (4->32) in VALU: lane computes 16 outputs of edge ln ----
            PK16 pk;
#pragma unroll
            for (int jj = 0; jj < 16; ++jj) {
                const int j = hi * 16 + jj;
                float s = sb1[j];
                s = fmaf(p0, sW1[0 * H_DIM + j], s);
                s = fmaf(p1, sW1[1 * H_DIM + j], s);
                s = fmaf(p2, sW1[2 * H_DIM + j], s);
                s = fmaf(p3, sW1[3 * H_DIM + j], s);
                s = fmaxf(s, 0.f);
                pk.h[jj] = (_Float16)s;
            }
            {
                int4* hd = (int4*)&hbase[ln * H_DIM + hi * 16];
                hd[0] = pk.q[0]; hd[1] = pk.q[1];
            }
            wave_lds_fence();

            // ---- layer 2 (32->32) via v_wmma_f32_16x16x32_f16 ----
            // A layout (16-bit 16x32): lane<16 -> K 0..7 & 16..23 of row ln; lane>=16 -> K 8..15 & 24..31
            ABV a;
            const _Float16* hrow = hbase + ln * H_DIM;
            a.q[0] = *(const int4*)(hrow + hi * 8);
            a.q[1] = *(const int4*)(hrow + 16 + hi * 8);

            v8f h2c[2];
#pragma unroll
            for (int t = 0; t < 2; ++t) {
                ABV b;
                const _Float16* wc = sW2T + (t * 16 + ln) * H_DIM + hi * 8;
                b.q[0] = *(const int4*)(wc);
                b.q[1] = *(const int4*)(wc + 16);
                v8f z = {};
                h2c[t] = __builtin_amdgcn_wmma_f32_16x16x32_f16(
                    false, a.v, false, b.v, (short)0, z, false, false);
            }
            // bias + relu on C layout (VGPR p -> row M = p + 8*hi, lane -> col), write h2 back (f16)
#pragma unroll
            for (int t = 0; t < 2; ++t) {
                const float bb = sb2[t * 16 + ln];
#pragma unroll
                for (int p = 0; p < 8; ++p) {
                    const float v = fmaxf(h2c[t][p] + bb, 0.f);
                    hbase[(p + hi * 8) * H_DIM + t * 16 + ln] = (_Float16)v;
                }
            }
            wave_lds_fence();

            // ---- layer 3 (32->256) + einsum fold ----
            ABV a3;
            a3.q[0] = *(const int4*)(hrow + hi * 8);
            a3.q[1] = *(const int4*)(hrow + 16 + hi * 8);
            const float* frow0 = fbase + (hi * 8) * IN_DIM;  // rows M = hi*8 .. hi*8+7
#pragma unroll
            for (int i = 0; i < IN_DIM; ++i) {
                ABV b;
                const _Float16* wc = sW3T + (i * 16 + ln) * H_DIM + hi * 8;
                b.q[0] = *(const int4*)(wc);
                b.q[1] = *(const int4*)(wc + 16);
                v8f z = {};
                v8f dmat = __builtin_amdgcn_wmma_f32_16x16x32_f16(
                    false, a3.v, false, b.v, (short)0, z, false, false);
                const float bias = sb3[i * 16 + ln];
#pragma unroll
                for (int p = 0; p < 8; ++p) {
                    // edge_out[M][o=ln] += feats[M][i] * w[M][i][o]
                    acc[p] = fmaf(frow0[p * IN_DIM + i], dmat[p] + bias, acc[p]);
                }
            }
        }

        // ---- segment sum over the 16 edge rows ----
        float s = 0.f;
#pragma unroll
        for (int p = 0; p < 8; ++p) s += acc[p];
        s += __shfl_xor(s, 16);   // combine wave halves (wave32)
        if (hi == 0) out[(size_t)q * OUT_DIM + ln] = s;
    }
}

extern "C" void kernel_launch(void* const* d_in, const int* in_sizes, int n_in,
                              void* d_out, int out_size, void* d_ws, size_t ws_size,
                              hipStream_t stream) {
    const float* feats   = (const float*)d_in[0];
    const float* xyz     = (const float*)d_in[1];
    const int*   nidx    = (const int*)d_in[2];
    const int*   rsplits = (const int*)d_in[3];
    const float* ndist   = (const float*)d_in[4];
    const float* W1      = (const float*)d_in[5];
    const float* b1      = (const float*)d_in[6];
    const float* W2      = (const float*)d_in[7];
    const float* b2      = (const float*)d_in[8];
    const float* W3      = (const float*)d_in[9];
    const float* b3      = (const float*)d_in[10];
    float* out = (float*)d_out;

    const int nq = in_sizes[3] - 1;
    int blocks = (nq + NWAVES_PER_BLOCK - 1) / NWAVES_PER_BLOCK;
    if (blocks > 2560) blocks = 2560;   // persistent waves: amortize LDS weight staging
    if (blocks < 1) blocks = 1;

    polarconv_kernel<<<blocks, 32 * NWAVES_PER_BLOCK, 0, stream>>>(
        feats, xyz, nidx, rsplits, ndist, W1, b1, W2, b2, W3, b3, out, nq);
}
